// ChunkedCrossAttention_35905926595227
// MI455X (gfx1250) — matrix operand order repaired and checked by reading, not compiled, not run on hardware
//
#include <hip/hip_runtime.h>

// ---------------------------------------------------------------------------
// ChunkedCrossAttention for MI455X (gfx1250, wave32, WMMA bf16)
// Layout strategy: weights pre-transposed, V written transposed by its GEMM
// epilogue -> every WMMA A/B fragment is contiguous 16B/32B global or LDS
// reads; no runtime scalar transposition on the hot path.
// ---------------------------------------------------------------------------
typedef __bf16 bf16_t;
typedef __attribute__((ext_vector_type(16))) __bf16 v16bf;
typedef __attribute__((ext_vector_type(8)))  float  v8f;

#define D_MODEL      1024
#define N_HEADS      16
#define D_K          64
#define CHUNK_LEN    64
#define BATCH        8
#define CHUNKS       32
#define KV_LEN       256                        // NEIGHBORS * NEIGHBOR_LEN
#define SEQ          (CHUNKS * CHUNK_LEN)       // 2048
#define ROWS_X       (BATCH * SEQ)              // 16384
#define ROWS_E       (BATCH * CHUNKS * KV_LEN)  // 65536
#define VALID_X      (SEQ - CHUNK_LEN + 1)      // 1985
#define PB_STRIDE    264                        // 256 + 8 pad (16B-aligned rows)

// ---------------------------------------------------------------------------
// f32 -> bf16 elementwise convert
// ---------------------------------------------------------------------------
__global__ __launch_bounds__(256) void cvt_bf16_kernel(const float* __restrict__ src,
                                                       bf16_t* __restrict__ dst,
                                                       size_t n) {
    size_t i = (size_t)blockIdx.x * blockDim.x + threadIdx.x;
    size_t stride = (size_t)gridDim.x * blockDim.x;
    for (; i < n; i += stride) dst[i] = (bf16_t)src[i];
}

// ---------------------------------------------------------------------------
// f32 KxN weight -> bf16 NxK (transposed) ; coalesced read, scattered write
// ---------------------------------------------------------------------------
__global__ __launch_bounds__(256) void cvt_wt_kernel(const float* __restrict__ W,
                                                     bf16_t* __restrict__ Wt) {
    size_t i = (size_t)blockIdx.x * blockDim.x + threadIdx.x;
    size_t stride = (size_t)gridDim.x * blockDim.x;
    const size_t n = (size_t)D_MODEL * D_MODEL;
    for (; i < n; i += stride) {
        const int k = (int)(i >> 10);
        const int c = (int)(i & (D_MODEL - 1));
        Wt[(size_t)c * D_MODEL + k] = (bf16_t)W[i];
    }
}

// ---------------------------------------------------------------------------
// LayerNorm of h[:, 63:] with zero padding to SEQ rows, output bf16
// ---------------------------------------------------------------------------
__global__ __launch_bounds__(256) void ln_shift_kernel(const float* __restrict__ h,
                                                       const float* __restrict__ gamma,
                                                       const float* __restrict__ beta,
                                                       bf16_t* __restrict__ x) {
    __shared__ float red[256];
    const int row = blockIdx.x;            // 0 .. ROWS_X-1
    const int b   = row >> 11;
    const int tp  = row & (SEQ - 1);
    bf16_t* xr = x + (size_t)row * D_MODEL;
    const int tid = threadIdx.x;

    if (tp >= VALID_X) {                   // reference pads AFTER LN -> zero rows
#pragma unroll
        for (int i = 0; i < 4; ++i) xr[tid + i * 256] = (bf16_t)0.0f;
        return;
    }
    const float* src = h + ((size_t)b * SEQ + tp + (CHUNK_LEN - 1)) * D_MODEL;

    float v[4];
    float s = 0.0f;
#pragma unroll
    for (int i = 0; i < 4; ++i) { v[i] = src[tid + i * 256]; s += v[i]; }
    red[tid] = s; __syncthreads();
    for (int o = 128; o > 0; o >>= 1) { if (tid < o) red[tid] += red[tid + o]; __syncthreads(); }
    const float mean = red[0] * (1.0f / D_MODEL);
    __syncthreads();

    float s2 = 0.0f;
#pragma unroll
    for (int i = 0; i < 4; ++i) { float d = v[i] - mean; s2 += d * d; }
    red[tid] = s2; __syncthreads();
    for (int o = 128; o > 0; o >>= 1) { if (tid < o) red[tid] += red[tid + o]; __syncthreads(); }
    const float rstd = rsqrtf(red[0] * (1.0f / D_MODEL) + 1e-5f);

#pragma unroll
    for (int i = 0; i < 4; ++i) {
        int idx = tid + i * 256;
        xr[idx] = (bf16_t)((v[i] - mean) * rstd * gamma[idx] + beta[idx]);
    }
}

// ---------------------------------------------------------------------------
// WMMA bf16 GEMM, no LDS:  C[MxN] = A[MxK] * Bt^T + bias
//   A  : M x K row-major bf16
//   Bt : N x K bf16 (B transposed) -> B fragments are 32B contiguous runs
// Block = 128 threads (4 waves); block tile 128x64; wave tile 32x64
// (8 WMMA per 32-wide K step vs 12 b128 loads).
// store_mode: 0 = bf16 row-major, 1 = f32 row-major, 2 = bf16 V-transposed
// ---------------------------------------------------------------------------
__global__ __launch_bounds__(128) void gemm_bf16_kernel(const bf16_t* __restrict__ A,
                                                        const bf16_t* __restrict__ Bt,
                                                        const float* __restrict__ bias,
                                                        bf16_t* __restrict__ Cb,
                                                        float* __restrict__ Cf,
                                                        int M, int N, int K,
                                                        int store_mode) {
    const int lane = threadIdx.x & 31;
    const int wave = threadIdx.x >> 5;
    const int bm   = blockIdx.x * 128 + wave * 32;
    const int bn   = blockIdx.y * 64;
    const int nl   = lane & 15;
    const int half = lane >> 4;

    v8f acc[2][4];
#pragma unroll
    for (int s = 0; s < 2; ++s)
#pragma unroll
        for (int t = 0; t < 4; ++t)
#pragma unroll
            for (int r = 0; r < 8; ++r) acc[s][t][r] = 0.0f;

    const bf16_t* Arow0 = A + (size_t)(bm + nl) * K;
    const bf16_t* Arow1 = Arow0 + (size_t)16 * K;

    for (int k0 = 0; k0 < K; k0 += 32) {
        if (k0 + 256 < K) __builtin_prefetch(Arow0 + k0 + 256, 0, 1);

        v16bf a0, a1;
        const bf16_t* pa0 = Arow0 + k0 + half * 16;
        const bf16_t* pa1 = Arow1 + k0 + half * 16;
#pragma unroll
        for (int i = 0; i < 8; ++i) {
            a0[i] = pa0[i]; a0[8 + i] = pa0[8 + i];
            a1[i] = pa1[i]; a1[8 + i] = pa1[8 + i];
        }
        v16bf bfr[4];
#pragma unroll
        for (int nt = 0; nt < 4; ++nt) {
            const bf16_t* pb = Bt + (size_t)(bn + nt * 16 + nl) * K + k0 + half * 16;
#pragma unroll
            for (int i = 0; i < 16; ++i) bfr[nt][i] = pb[i];
        }
#pragma unroll
        for (int nt = 0; nt < 4; ++nt) {
            acc[0][nt] = __builtin_amdgcn_wmma_f32_16x16x32_bf16(
                false, a0, false, bfr[nt], (short)0, acc[0][nt], false, false);
            acc[1][nt] = __builtin_amdgcn_wmma_f32_16x16x32_bf16(
                false, a1, false, bfr[nt], (short)0, acc[1][nt], false, false);
        }
    }

    // epilogue: C layout m = r + 8*half (+16*sub), n = lane&15
#pragma unroll
    for (int s = 0; s < 2; ++s) {
#pragma unroll
        for (int nt = 0; nt < 4; ++nt) {
            const int n = bn + nt * 16 + nl;
            const float bi = bias ? bias[n] : 0.0f;
#pragma unroll
            for (int r = 0; r < 8; ++r) {
                const int m = bm + s * 16 + r + 8 * half;
                const float val = acc[s][nt][r] + bi;
                if (store_mode == 0) {
                    Cb[(size_t)m * N + n] = (bf16_t)val;
                } else if (store_mode == 1) {
                    Cf[(size_t)m * N + n] = val;
                } else {
                    // V transposed: [bc][head][d][j], bc = m>>8, j = m&255
                    const int bc = m >> 8, j = m & 255;
                    const int hh = n >> 6, d = n & 63;
                    Cb[(((size_t)bc * N_HEADS + hh) * D_K + d) * KV_LEN + j] = (bf16_t)val;
                }
            }
        }
    }
}

// ---------------------------------------------------------------------------
// Attention: one block (4 waves) per (b, chunk, head); waves are independent
// (no barriers).  Wave w owns score rows 16w..16w+15.
//   Q fragments  : contiguous global (row-major q)
//   K fragments  : contiguous global (B matrix = K^T, lane indexes K's row)
//   P            : LDS per-wave, padded stride (softmax lane-transpose)
//   V fragments  : contiguous global from pre-transposed vt buffer
// ---------------------------------------------------------------------------
__global__ __launch_bounds__(128) void attn_kernel(const bf16_t* __restrict__ q,
                                                   const bf16_t* __restrict__ k,
                                                   const bf16_t* __restrict__ vt,
                                                   bf16_t* __restrict__ o) {
    __shared__ bf16_t Pb[4][16][PB_STRIDE];         // ~33 KB

    const int blk = blockIdx.x;
    const int hh  = blk & (N_HEADS - 1);
    const int c   = (blk >> 4) & (CHUNKS - 1);
    const int b   = blk >> 9;
    const int bc  = b * CHUNKS + c;

    const bf16_t* Q   = q  + ((size_t)(b * SEQ + c * CHUNK_LEN)) * D_MODEL + hh * D_K;
    const bf16_t* Kp  = k  + ((size_t)bc * KV_LEN) * D_MODEL + hh * D_K;
    const bf16_t* Vtp = vt + ((size_t)bc * N_HEADS + hh) * D_K * KV_LEN;

    const int lane = threadIdx.x & 31;
    const int wave = threadIdx.x >> 5;
    const int nl   = lane & 15;
    const int half = lane >> 4;

    // ---- scores: Q * K^T -------------------------------------------------
    v8f sc[16];
#pragma unroll
    for (int t = 0; t < 16; ++t)
#pragma unroll
        for (int r = 0; r < 8; ++r) sc[t][r] = 0.0f;

    v16bf aq[2];
#pragma unroll
    for (int kk = 0; kk < 2; ++kk) {
        const bf16_t* pa = Q + (size_t)(wave * 16 + nl) * D_MODEL + kk * 32 + half * 16;
#pragma unroll
        for (int i = 0; i < 8; ++i) { aq[kk][i] = pa[i]; aq[kk][8 + i] = pa[8 + i]; }
    }
#pragma unroll
    for (int jt = 0; jt < 16; ++jt) {
#pragma unroll
        for (int kk = 0; kk < 2; ++kk) {
            v16bf bk;
            const bf16_t* pb = Kp + (size_t)(jt * 16 + nl) * D_MODEL + kk * 32 + half * 16;
#pragma unroll
            for (int i = 0; i < 16; ++i) bk[i] = pb[i];
            sc[jt] = __builtin_amdgcn_wmma_f32_16x16x32_bf16(
                false, aq[kk], false, bk, (short)0, sc[jt], false, false);
        }
    }

    // ---- softmax over 256 (scale = 1/sqrt(64)) ---------------------------
    const float scale = 0.125f;
#pragma unroll
    for (int r = 0; r < 8; ++r) {
        float mx = -3.0e38f;
#pragma unroll
        for (int jt = 0; jt < 16; ++jt) {
            sc[jt][r] *= scale;
            mx = fmaxf(mx, sc[jt][r]);
        }
#pragma unroll
        for (int off = 1; off < 16; off <<= 1) mx = fmaxf(mx, __shfl_xor(mx, off, 32));
        float sum = 0.0f;
#pragma unroll
        for (int jt = 0; jt < 16; ++jt) {
            float p = __expf(sc[jt][r] - mx);
            sc[jt][r] = p;
            sum += p;
        }
#pragma unroll
        for (int off = 1; off < 16; off <<= 1) sum += __shfl_xor(sum, off, 32);
        const float inv = 1.0f / sum;
        const int m = r + 8 * half;
#pragma unroll
        for (int jt = 0; jt < 16; ++jt)
            Pb[wave][m][jt * 16 + nl] = (bf16_t)(sc[jt][r] * inv);
    }
    // same-wave LDS RAW: DS ops from one wave are in order; no barrier needed.

    // ---- O = P * V (V fragments contiguous from transposed vt) -----------
    v8f oa[4];
#pragma unroll
    for (int t = 0; t < 4; ++t)
#pragma unroll
        for (int r = 0; r < 8; ++r) oa[t][r] = 0.0f;

#pragma unroll
    for (int kc = 0; kc < KV_LEN / 32; ++kc) {
        v16bf a;
        const bf16_t* pa = &Pb[wave][nl][kc * 32 + half * 16];
#pragma unroll
        for (int i = 0; i < 8; ++i) { a[i] = pa[i]; a[8 + i] = pa[8 + i]; }
#pragma unroll
        for (int nt = 0; nt < 4; ++nt) {
            v16bf bv;
            const bf16_t* pb = Vtp + (size_t)(nt * 16 + nl) * KV_LEN + kc * 32 + half * 16;
#pragma unroll
            for (int i = 0; i < 16; ++i) bv[i] = pb[i];
            oa[nt] = __builtin_amdgcn_wmma_f32_16x16x32_bf16(
                false, a, false, bv, (short)0, oa[nt], false, false);
        }
    }

    // ---- store O (bf16, heads concatenated, row-major) -------------------
#pragma unroll
    for (int nt = 0; nt < 4; ++nt) {
#pragma unroll
        for (int r = 0; r < 8; ++r) {
            const int row = b * SEQ + c * CHUNK_LEN + wave * 16 + r + 8 * half;
            const int col = hh * D_K + nt * 16 + nl;
            o[(size_t)row * D_MODEL + col] = (bf16_t)oa[nt][r];
        }
    }
}

// ---------------------------------------------------------------------------
// out[b,t] = h[b,t] + (t >= 63 ? oproj[b, t-63] : 0)
// ---------------------------------------------------------------------------
__global__ __launch_bounds__(256) void residual_shift_kernel(const float* __restrict__ h,
                                                             const float* __restrict__ oproj,
                                                             float* __restrict__ out,
                                                             size_t n) {
    size_t i = (size_t)blockIdx.x * blockDim.x + threadIdx.x;
    size_t stride = (size_t)gridDim.x * blockDim.x;
    for (; i < n; i += stride) {
        const int    col = (int)(i & (D_MODEL - 1));
        const size_t row = i >> 10;
        const int    t   = (int)(row & (SEQ - 1));
        const size_t b   = row >> 11;
        float r = h[i];
        if (t >= CHUNK_LEN - 1)
            r += oproj[((b << 11) + (size_t)(t - (CHUNK_LEN - 1))) * D_MODEL + col];
        out[i] = r;
    }
}

// ---------------------------------------------------------------------------
// Launch
// ---------------------------------------------------------------------------
extern "C" void kernel_launch(void* const* d_in, const int* in_sizes, int n_in,
                              void* d_out, int out_size, void* d_ws, size_t ws_size,
                              hipStream_t stream) {
    (void)in_sizes; (void)n_in; (void)out_size; (void)ws_size;
    const float* h     = (const float*)d_in[0];
    const float* e     = (const float*)d_in[1];
    const float* Wq    = (const float*)d_in[2];
    const float* bq    = (const float*)d_in[3];
    const float* Wk    = (const float*)d_in[4];
    const float* bk    = (const float*)d_in[5];
    const float* Wv    = (const float*)d_in[6];
    const float* bv    = (const float*)d_in[7];
    const float* gamma = (const float*)d_in[8];
    const float* beta  = (const float*)d_in[9];
    const float* Wo    = (const float*)d_in[10];
    const float* bo    = (const float*)d_in[11];
    float* out = (float*)d_out;

    char* ws = (char*)d_ws;
    size_t off = 0;
    auto carve = [&](size_t bytes) -> char* {
        char* p = ws + off;
        off += (bytes + 255) & ~(size_t)255;
        return p;
    };
    bf16_t* xb    = (bf16_t*)carve((size_t)ROWS_X * D_MODEL * 2);
    bf16_t* eb    = (bf16_t*)carve((size_t)ROWS_E * D_MODEL * 2);
    bf16_t* wqt   = (bf16_t*)carve((size_t)D_MODEL * D_MODEL * 2);
    bf16_t* wkt   = (bf16_t*)carve((size_t)D_MODEL * D_MODEL * 2);
    bf16_t* wvt   = (bf16_t*)carve((size_t)D_MODEL * D_MODEL * 2);
    bf16_t* wot   = (bf16_t*)carve((size_t)D_MODEL * D_MODEL * 2);
    bf16_t* qb    = (bf16_t*)carve((size_t)ROWS_X * D_MODEL * 2);
    bf16_t* kb    = (bf16_t*)carve((size_t)ROWS_E * D_MODEL * 2);
    bf16_t* vtb   = (bf16_t*)carve((size_t)ROWS_E * D_MODEL * 2);
    bf16_t* ob    = (bf16_t*)carve((size_t)ROWS_X * D_MODEL * 2);
    float*  oproj = (float*) carve((size_t)ROWS_X * D_MODEL * 4);

    // 1) weights -> bf16 transposed ; e -> bf16
    cvt_wt_kernel<<<512, 256, 0, stream>>>(Wq, wqt);
    cvt_wt_kernel<<<512, 256, 0, stream>>>(Wk, wkt);
    cvt_wt_kernel<<<512, 256, 0, stream>>>(Wv, wvt);
    cvt_wt_kernel<<<512, 256, 0, stream>>>(Wo, wot);
    cvt_bf16_kernel<<<8192, 256, 0, stream>>>(e, eb, (size_t)ROWS_E * D_MODEL);

    // 2) LayerNorm + shift + pad -> x (bf16)
    ln_shift_kernel<<<ROWS_X, 256, 0, stream>>>(h, gamma, beta, xb);

    // 3) projections (LDS-free WMMA GEMMs, weights transposed)
    gemm_bf16_kernel<<<dim3(ROWS_X / 128, D_MODEL / 64), 128, 0, stream>>>(
        xb, wqt, bq, qb, nullptr, ROWS_X, D_MODEL, D_MODEL, 0);
    gemm_bf16_kernel<<<dim3(ROWS_E / 128, D_MODEL / 64), 128, 0, stream>>>(
        eb, wkt, bk, kb, nullptr, ROWS_E, D_MODEL, D_MODEL, 0);
    gemm_bf16_kernel<<<dim3(ROWS_E / 128, D_MODEL / 64), 128, 0, stream>>>(
        eb, wvt, bv, vtb, nullptr, ROWS_E, D_MODEL, D_MODEL, 2);   // V transposed

    // 4) attention per (b, chunk, head)
    attn_kernel<<<BATCH * CHUNKS * N_HEADS, 128, 0, stream>>>(qb, kb, vtb, ob);

    // 5) output projection (f32) + residual/shift epilogue
    gemm_bf16_kernel<<<dim3(ROWS_X / 128, D_MODEL / 64), 128, 0, stream>>>(
        ob, wot, bo, nullptr, oproj, ROWS_X, D_MODEL, D_MODEL, 1);
    residual_shift_kernel<<<8192, 256, 0, stream>>>(h, oproj, out,
                                                    (size_t)ROWS_X * D_MODEL);
}
